// H_Encoder_17446157156468
// MI455X (gfx1250) — compile-verified, matrix operand order checked by reading, not run on hardware
//
#include <hip/hip_runtime.h>
#include <hip/hip_bf16.h>
#include <math.h>

// ---------------------------------------------------------------------------
// MI455X (gfx1250) implementation of the Lorentz hyperbolic encoder.
//
// Roofline: the two 513x65537 fp32 heads stream ~269MB of weights -> HBM bound
// (~11.5us at 23.3TB/s); handled by a streaming dot kernel + global_prefetch.
// The four conv blocks are GEMM-shaped (K up to 1153) -> v_wmma_f32_16x16x32_f16
// with im2col + Lorentz time-fix fused into LDS staging. The LDS tiles are
// stored pre-swizzled into the ISA 7.12.2 per-lane fragment layout so each
// wave's A/B fragments load as two aligned ds_load_b128 each (no half-gather).
// All Lorentz BN math runs as wave32 wave-per-row kernels with shfl reductions.
//
// Workspace layout (floats): buf0 [0,18M), buf1 [18M,36M), reductions at 36M.
// Requires ws_size >= ~145MB.
// ---------------------------------------------------------------------------

typedef __attribute__((ext_vector_type(16))) _Float16 v16h;
typedef __attribute__((ext_vector_type(8)))  float    v8f;

#define WS_BUF_FLOATS (18u * 1024u * 1024u)

__device__ __forceinline__ float waveRedSum(float v) {
#pragma unroll
  for (int off = 16; off > 0; off >>= 1) v += __shfl_xor(v, off, 32);
  return v;
}

// ---------------- init: NCHW fp32 -> (B,H,W,4) Lorentz points ----------------
__global__ void k_init(const float* __restrict__ x, float* __restrict__ h,
                       int B, int HW) {
  int idx = blockIdx.x * blockDim.x + threadIdx.x;
  if (idx >= B * HW) return;
  int b = idx / HW, p = idx % HW;
  float v1 = x[((size_t)b * 3 + 0) * HW + p];
  float v2 = x[((size_t)b * 3 + 1) * HW + p];
  float v3 = x[((size_t)b * 3 + 2) * HW + p];
  float* o = h + (size_t)idx * 4;
  o[0] = sqrtf(v1 * v1 + v2 * v2 + v3 * v3 + 1.0f);
  o[1] = v1; o[2] = v2; o[3] = v3;
}

// Inverse of the ISA 7.12.2 16-bit A/B fragment layout:
// a lane's 16 halves cover k in {base..base+7, base+16..base+23}, base=8*(lane>=16).
__device__ __forceinline__ int fragSlot(int k) {
  return (k < 16) ? (k & 7) : ((k & 7) + 8);
}
__device__ __forceinline__ int fragHalf(int k) {  // 0 -> lanes 0-15, 1 -> 16-31
  return ((k & 15) >= 8) ? 1 : 0;
}

// ---------------- conv block GEMM: im2col+time-fix fused, WMMA f16 ----------
// in : (B,Hin,Win,Cin) fp32 Lorentz points
// W  : (Cout, Kdim) fp32, Kdim = (Cin-1)*9+1
// out: lin (M, Cout), M = B*(Hin/2)*(Win/2); column 0 fixed up later.
__global__ __launch_bounds__(128)
void k_conv_gemm(const float* __restrict__ in, const float* __restrict__ W,
                 const float* __restrict__ bias, float* __restrict__ out,
                 int Batch, int Cin, int Hin, int Win, int Cout) {
  const int Hout = Hin >> 1, Wout = Win >> 1;
  const int Npix = Hout * Wout;
  const int M = Batch * Npix;
  const int Kdim = (Cin - 1) * 9 + 1;
  const int N = Cout;

  // Fragment-swizzled tiles: [lane][slot], 32B per lane -> ds_load_b128 x2.
  __shared__ __align__(32) _Float16 As[32][16];      // 16M x 32K tile
  __shared__ __align__(32) _Float16 Bs[4][32][16];   // 32K x 64N tile (4 waves)

  const int tid  = threadIdx.x;
  const int wave = tid >> 5;
  const int lane = tid & 31;
  const int mt = blockIdx.x;        // 16-row M tile
  const int nBaseBlk = blockIdx.y * 64;

  v8f acc = {};

  for (int k0 = 0; k0 < Kdim; k0 += 32) {
    // ---- stage A (im2col with Lorentz time merge at j==0), swizzled ----
    for (int e = tid; e < 16 * 32; e += 128) {
      int ml = e >> 5, kk = e & 31;
      int m = mt * 16 + ml;
      int j = k0 + kk;
      float val = 0.0f;
      if (m < M && j < Kdim) {
        int b = m / Npix, pp = m % Npix;
        int ho = pp / Wout, wo = pp % Wout;
        if (j == 0) {
          // t_new = sqrt(sum_k max(t_k,1)^2 - 8K); zero-padded taps -> max(0,1)=1
          float s = 0.0f;
#pragma unroll
          for (int kp = 0; kp < 9; ++kp) {
            int hi = 2 * ho - 1 + kp / 3, wi = 2 * wo - 1 + kp % 3;
            float tv = 0.0f;
            if (hi >= 0 && hi < Hin && wi >= 0 && wi < Win)
              tv = in[(((size_t)b * Hin + hi) * Win + wi) * Cin + 0];
            tv = fmaxf(tv, 1.0f);
            s += tv * tv;
          }
          val = sqrtf(fmaxf(s - 8.0f, 0.0f));
        } else {
          int c = 1 + (j - 1) / 9, kp = (j - 1) % 9;
          int hi = 2 * ho - 1 + kp / 3, wi = 2 * wo - 1 + kp % 3;
          if (hi >= 0 && hi < Hin && wi >= 0 && wi < Win)
            val = in[(((size_t)b * Hin + hi) * Win + wi) * Cin + c];
        }
      }
      As[ml + 16 * fragHalf(kk)][fragSlot(kk)] = (_Float16)val;
    }
    // ---- stage B (weights, K-major), swizzled per wave sub-tile ----
    for (int e = tid; e < 32 * 64; e += 128) {
      int kk = e >> 6, nl = e & 63;
      int j = k0 + kk, n = nBaseBlk + nl;
      float val = (j < Kdim && n < N) ? W[(size_t)n * Kdim + j] : 0.0f;
      Bs[nl >> 4][(nl & 15) + 16 * fragHalf(kk)][fragSlot(kk)] = (_Float16)val;
    }
    __syncthreads();

    // ---- fragments are contiguous per lane: vector LDS loads ----
    v16h a  = *(const v16h*)&As[lane][0];
    v16h bf = *(const v16h*)&Bs[wave][lane][0];
    acc = __builtin_amdgcn_wmma_f32_16x16x32_f16(
        false, a, false, bf, (short)0, acc, false, false);
    __syncthreads();
  }

  // ---- epilogue: D layout (VGPR j -> M=j+8*(lane>=16), N=lane&15) ----
  const int n = nBaseBlk + wave * 16 + (lane & 15);
#pragma unroll
  for (int j = 0; j < 8; ++j) {
    int m = mt * 16 + j + ((lane < 16) ? 0 : 8);
    if (m < M && n < N)
      out[(size_t)m * N + n] = acc[j] + bias[n];
  }
}

// ---------------- add_time on each GEMM row, in place ------------------------
__global__ void k_row_time(float* __restrict__ y, int M, int C) {
  int wid  = (blockIdx.x * blockDim.x + threadIdx.x) >> 5;
  int lane = threadIdx.x & 31;
  if (wid >= M) return;
  float* row = y + (size_t)wid * C;
  float s = 0.0f;
  for (int c = 1 + lane; c < C; c += 32) { float v = row[c]; s += v * v; }
  s = waveRedSum(s);
  if (lane == 0) row[0] = sqrtf(s + 1.0f);
}

__global__ void k_zero(float* __restrict__ p, int n) {
  int i = blockIdx.x * blockDim.x + threadIdx.x;
  if (i < n) p[i] = 0.0f;
}

// ---------------- BN stage 1: per-batch channel sums -------------------------
__global__ void k_bn_sums(const float* __restrict__ y, float* __restrict__ sums,
                          int Npix, int C, int chunks) {
  extern __shared__ float sacc[];  // C floats
  int b = blockIdx.x, ch = blockIdx.y;
  for (int c = threadIdx.x; c < C; c += blockDim.x) sacc[c] = 0.0f;
  __syncthreads();
  int per = (Npix + chunks - 1) / chunks;
  int r0 = ch * per, r1 = min(Npix, r0 + per);
  for (int r = r0; r < r1; ++r) {
    const float* row = y + ((size_t)b * Npix + r) * C;
    for (int c = threadIdx.x; c < C; c += blockDim.x) sacc[c] += row[c];
  }
  __syncthreads();
  for (int c = threadIdx.x; c < C; c += blockDim.x)
    atomicAdd(&sums[b * C + c], sacc[c]);
}

// ---------------- BN stage 2: double Lorentz centroid -> mean (C,) -----------
__global__ void k_bn_mean(const float* __restrict__ sums,
                          float* __restrict__ meanvec,
                          int B, int Npix, int C) {
  __shared__ float c1[32][258];
  __shared__ float avg2[258];
  __shared__ float dnm;
  int tid = threadIdx.x;
  if (tid < B) {
    float inv = 1.0f / (float)Npix;
    float inner = 0.0f;
    for (int c = 0; c < C; ++c) {
      float a = sums[tid * C + c] * inv;
      c1[tid][c] = a;
      inner += (c == 0) ? -a * a : a * a;
    }
    float r = 1.0f / sqrtf(fmaxf(fabsf(inner), 1e-8f));
    for (int c = 0; c < C; ++c) c1[tid][c] *= r;
  }
  __syncthreads();
  for (int c = tid; c < C; c += blockDim.x) {
    float s = 0.0f;
    for (int b = 0; b < B; ++b) s += c1[b][c];
    avg2[c] = s / (float)B;
  }
  __syncthreads();
  if (tid == 0) {
    float inner = 0.0f;
    for (int c = 0; c < C; ++c)
      inner += (c == 0) ? -avg2[c] * avg2[c] : avg2[c] * avg2[c];
    dnm = 1.0f / sqrtf(fmaxf(fabsf(inner), 1e-8f));
  }
  __syncthreads();
  for (int c = tid; c < C; c += blockDim.x) meanvec[c] = avg2[c] * dnm;
}

// ---------------- BN stage 3: Frechet variance (sum of ||u||) ----------------
__global__ void k_bn_var(const float* __restrict__ y,
                         const float* __restrict__ meanvec,
                         float* __restrict__ varsum, int M, int C) {
  extern __shared__ float smean[];
  for (int c = threadIdx.x; c < C; c += blockDim.x) smean[c] = meanvec[c];
  __syncthreads();
  int wid  = (blockIdx.x * blockDim.x + threadIdx.x) >> 5;
  int lane = threadIdx.x & 31;
  if (wid >= M) return;
  const float* row = y + (size_t)wid * C;

  float xy = 0.0f;
  for (int c = lane; c < C; c += 32) {
    float t = smean[c] * row[c];
    xy += (c == 0) ? -t : t;
  }
  xy = waveRedSum(xy);
  float dist = acoshf(fmaxf(-xy, 1.0f + 1e-7f));

  float inner = 0.0f;
  for (int c = lane; c < C; c += 32) {
    float nm = row[c] + xy * smean[c];
    inner += (c == 0) ? -nm * nm : nm * nm;
  }
  inner = waveRedSum(inner);
  float den = sqrtf(fmaxf(inner, 1e-8f));
  float n0 = row[0] + xy * smean[0];
  float u0 = dist * n0 / den;
  float coefT = -u0 / (1.0f + smean[0]);

  float nrm = 0.0f;
  for (int c = lane; c < C; c += 32) {
    float nm = row[c] + xy * smean[c];
    float u = dist * nm / den + coefT * (smean[c] + ((c == 0) ? 1.0f : 0.0f));
    nrm += u * u;
  }
  nrm = waveRedSum(nrm);
  if (lane == 0) atomicAdd(varsum, sqrtf(nrm));
}

// ---------------- BN stage 4: rescale, transp0, expmap, relu, add_time -------
__global__ void k_bn_apply(const float* __restrict__ y,
                           const float* __restrict__ meanvec,
                           const float* __restrict__ varsum,
                           const float* __restrict__ beta,
                           const float* __restrict__ gamma,
                           float* __restrict__ out, int M, int C) {
  extern __shared__ float sh2[];  // 2*C floats
  float* smean = sh2;
  float* sbeta = sh2 + C;
  for (int c = threadIdx.x; c < C; c += blockDim.x) {
    smean[c] = meanvec[c];
    sbeta[c] = beta[c];
  }
  __syncthreads();
  int wid  = (blockIdx.x * blockDim.x + threadIdx.x) >> 5;
  int lane = threadIdx.x & 31;
  if (wid >= M) return;
  const float* row = y + (size_t)wid * C;
  float* orow = out + (size_t)wid * C;

  float var = varsum[0] / (float)M;
  float g = gamma[0] / (var + 1e-5f);

  float xy = 0.0f;
  for (int c = lane; c < C; c += 32) {
    float t = smean[c] * row[c];
    xy += (c == 0) ? -t : t;
  }
  xy = waveRedSum(xy);
  float dist = acoshf(fmaxf(-xy, 1.0f + 1e-7f));

  float inner = 0.0f;
  for (int c = lane; c < C; c += 32) {
    float nm = row[c] + xy * smean[c];
    inner += (c == 0) ? -nm * nm : nm * nm;
  }
  inner = waveRedSum(inner);
  float den = sqrtf(fmaxf(inner, 1e-8f));
  float n0 = row[0] + xy * smean[0];
  float u0 = dist * n0 / den;
  float coefT = -u0 / (1.0f + smean[0]);

  // transp0(beta, u'): coef2 = l_inner(beta, u') / (1 + beta0)
  float sbu = 0.0f;
  for (int c = lane; c < C; c += 32) {
    float nm = row[c] + xy * smean[c];
    float u = g * (dist * nm / den + coefT * (smean[c] + ((c == 0) ? 1.0f : 0.0f)));
    float t = sbeta[c] * u;
    sbu += (c == 0) ? -t : t;
  }
  sbu = waveRedSum(sbu);
  float coef2 = sbu / (1.0f + sbeta[0]);

  float nn = 0.0f;
  for (int c = lane; c < C; c += 32) {
    float nm = row[c] + xy * smean[c];
    float uf = g * (dist * nm / den + coefT * (smean[c] + ((c == 0) ? 1.0f : 0.0f)))
             + coef2 * (sbeta[c] + ((c == 0) ? 1.0f : 0.0f));
    nn += (c == 0) ? -uf * uf : uf * uf;
  }
  nn = waveRedSum(nn);
  float nu = sqrtf(fmaxf(nn, 1e-8f));
  float chh = coshf(nu), shn = sinhf(nu) / nu;

  float ss = 0.0f;
  for (int c = lane; c < C; c += 32) {
    float nm = row[c] + xy * smean[c];
    float uf = g * (dist * nm / den + coefT * (smean[c] + ((c == 0) ? 1.0f : 0.0f)))
             + coef2 * (sbeta[c] + ((c == 0) ? 1.0f : 0.0f));
    float o = chh * sbeta[c] + shn * uf;
    if (c >= 1) { o = fmaxf(o, 0.0f); orow[c] = o; ss += o * o; }
  }
  ss = waveRedSum(ss);
  if (lane == 0) orow[0] = sqrtf(ss + 1.0f);
}

// ---------------- Lorentz flatten: (B,256,257) -> (B,65537) ------------------
__global__ void k_flatten(const float* __restrict__ x, float* __restrict__ h,
                          int Npix, int Cs) {
  int b = blockIdx.x;
  __shared__ float red[256];
  float s = 0.0f;
  int total = Npix * Cs;
  for (int i = threadIdx.x; i < total; i += blockDim.x) {
    int p = i / Cs, c = i % Cs;
    float v = x[((size_t)b * Npix + p) * (Cs + 1) + 1 + c];
    h[(size_t)b * (total + 1) + 1 + i] = v;
    s += v * v;
  }
  red[threadIdx.x] = s;
  __syncthreads();
  for (int off = blockDim.x >> 1; off > 0; off >>= 1) {
    if ((int)threadIdx.x < off) red[threadIdx.x] += red[threadIdx.x + off];
    __syncthreads();
  }
  if (threadIdx.x == 0) h[(size_t)b * (total + 1)] = sqrtf(red[0] + 1.0f);
}

// ---------------- bandwidth-bound head: lin[b][o] = h[b].wm[o] + b[o] --------
// 269MB fp32 weights -> pure streaming; prefetch hints the GL2 stream.
__global__ __launch_bounds__(256)
void k_final_gemm(const float* __restrict__ h, const float* __restrict__ W,
                  const float* __restrict__ bias, float* __restrict__ lin,
                  int Kd, int N) {
  int o = blockIdx.x;
  const float* wrow = W + (size_t)o * Kd;
  float acc[32];
#pragma unroll
  for (int b = 0; b < 32; ++b) acc[b] = 0.0f;
  for (int k = threadIdx.x; k < Kd; k += 256) {
    __builtin_prefetch(wrow + k + 4096, 0, 0);  // global_prefetch_b8
    float w = wrow[k];
#pragma unroll
    for (int b = 0; b < 32; ++b) acc[b] += w * h[(size_t)b * Kd + k];
  }
  __shared__ float red[256];
  for (int b = 0; b < 32; ++b) {
    red[threadIdx.x] = acc[b];
    __syncthreads();
    for (int off = 128; off > 0; off >>= 1) {
      if ((int)threadIdx.x < off) red[threadIdx.x] += red[threadIdx.x + off];
      __syncthreads();
    }
    if (threadIdx.x == 0) lin[(size_t)b * N + o] = red[0] + bias[o];
    __syncthreads();
  }
}

// ---------------- output epilogue: add_time(mean), softplus(var) -------------
__global__ void k_final_out(const float* __restrict__ linM,
                            const float* __restrict__ linV,
                            float* __restrict__ outp, int B, int N) {
  int b = blockIdx.x;
  __shared__ float red[256];
  float s = 0.0f;
  for (int o = 1 + threadIdx.x; o < N; o += blockDim.x) {
    float v = linM[(size_t)b * N + o];
    s += v * v;
  }
  red[threadIdx.x] = s;
  __syncthreads();
  for (int off = 128; off > 0; off >>= 1) {
    if ((int)threadIdx.x < off) red[threadIdx.x] += red[threadIdx.x + off];
    __syncthreads();
  }
  float* meanOut = outp;                  // (B, 513)
  float* varOut  = outp + (size_t)B * N;  // (B, 512)
  if (threadIdx.x == 0) meanOut[(size_t)b * N] = sqrtf(red[0] + 1.0f);
  for (int o = 1 + threadIdx.x; o < N; o += blockDim.x) {
    meanOut[(size_t)b * N + o] = linM[(size_t)b * N + o];
    float x = linV[(size_t)b * N + o];
    float sp = (x > 20.0f) ? x : log1pf(expf(x));
    varOut[(size_t)b * (N - 1) + (o - 1)] = fmaxf(sp, 1e-5f);
  }
}

// ---------------------------------------------------------------------------
extern "C" void kernel_launch(void* const* d_in, const int* in_sizes, int n_in,
                              void* d_out, int out_size, void* d_ws, size_t ws_size,
                              hipStream_t stream) {
  (void)in_sizes; (void)n_in; (void)out_size; (void)ws_size;
  const float* x = (const float*)d_in[0];

  float* ws   = (float*)d_ws;
  float* buf0 = ws;
  float* buf1 = ws + (size_t)WS_BUF_FLOATS;
  float* red  = ws + (size_t)2 * WS_BUF_FLOATS;
  float* sums  = red;          // 32*257 = 8224 floats
  float* meanv = red + 8224;   // up to 257 floats
  float* vsum  = red + 8481;   // 1 float
  float* linM  = red + 8704;   // 32*513
  float* linV  = linM + 16416; // 32*513

  const int B = 32;
  const int HW0 = 256 * 256;
  k_init<<<(B * HW0 + 255) / 256, 256, 0, stream>>>(x, buf0, B, HW0);

  int Cin = 4, Hin = 256, Win = 256;
  const int Couts[4] = {33, 65, 129, 257};
  for (int L = 0; L < 4; ++L) {
    const float* W     = (const float*)d_in[1 + 4 * L];
    const float* bconv = (const float*)d_in[2 + 4 * L];
    const float* beta  = (const float*)d_in[3 + 4 * L];
    const float* gamma = (const float*)d_in[4 + 4 * L];
    const int Cout = Couts[L];
    const int Hout = Hin / 2, Wout = Win / 2;
    const int Npix = Hout * Wout;
    const int M = B * Npix;

    dim3 g((M + 15) / 16, (Cout + 63) / 64);
    k_conv_gemm<<<g, 128, 0, stream>>>(buf0, W, bconv, buf1,
                                       B, Cin, Hin, Win, Cout);
    k_row_time<<<(M + 7) / 8, 256, 0, stream>>>(buf1, M, Cout);

    k_zero<<<(8482 + 255) / 256, 256, 0, stream>>>(red, 8482);
    k_bn_sums<<<dim3(B, 64), 256, Cout * sizeof(float), stream>>>(
        buf1, sums, Npix, Cout, 64);
    k_bn_mean<<<1, 288, 0, stream>>>(sums, meanv, B, Npix, Cout);
    k_bn_var<<<(M + 7) / 8, 256, Cout * sizeof(float), stream>>>(
        buf1, meanv, vsum, M, Cout);
    k_bn_apply<<<(M + 7) / 8, 256, 2 * Cout * sizeof(float), stream>>>(
        buf1, meanv, vsum, beta, gamma, buf0, M, Cout);

    Cin = Cout; Hin = Hout; Win = Wout;
  }

  // buf0 now holds (32, 16, 16, 257)
  k_flatten<<<32, 256, 0, stream>>>(buf0, buf1, 256, 256);

  const int Kd = 16 * 16 * 256 + 1;  // 65537
  const int Nf = 513;
  k_final_gemm<<<Nf, 256, 0, stream>>>(buf1, (const float*)d_in[17],
                                       (const float*)d_in[18], linM, Kd, Nf);
  k_final_gemm<<<Nf, 256, 0, stream>>>(buf1, (const float*)d_in[19],
                                       (const float*)d_in[20], linV, Kd, Nf);
  k_final_out<<<32, 256, 0, stream>>>(linM, linV, (float*)d_out, B, Nf);
}
